// WHNN_19851338842336
// MI455X (gfx1250) — matrix operand order repaired and checked by reading, not compiled, not run on hardware
//
#include <hip/hip_runtime.h>

#define N 4096
#define TOPK 409
#define LOG2N 12

typedef __attribute__((ext_vector_type(16))) __bf16 v16bf;
typedef __attribute__((ext_vector_type(8)))  float  v8f;

// ---------------------------------------------------------------------------
// Kernel 1: fcg[i][j] = max(1 - (|si-sj| - |ti-tj|)^2 / sigma^2, 0), diag = 0
// Block computes a 128x128 tile; point coords cached in LDS.
// ---------------------------------------------------------------------------
__global__ __launch_bounds__(256) void fcg_kernel(const float* __restrict__ src,
                                                  const float* __restrict__ tgt,
                                                  float* __restrict__ fcg) {
    __shared__ float si[128][3], ti[128][3], sj[128][3], tj[128][3];
    const int t = threadIdx.x;
    const int row0 = blockIdx.y * 128;
    const int col0 = blockIdx.x * 128;
    if (t < 128) {
        const int g = row0 + t;
        si[t][0] = src[g * 3 + 0]; si[t][1] = src[g * 3 + 1]; si[t][2] = src[g * 3 + 2];
        ti[t][0] = tgt[g * 3 + 0]; ti[t][1] = tgt[g * 3 + 1]; ti[t][2] = tgt[g * 3 + 2];
    } else {
        const int u = t - 128;
        const int g = col0 + u;
        sj[u][0] = src[g * 3 + 0]; sj[u][1] = src[g * 3 + 1]; sj[u][2] = src[g * 3 + 2];
        tj[u][0] = tgt[g * 3 + 0]; tj[u][1] = tgt[g * 3 + 1]; tj[u][2] = tgt[g * 3 + 2];
    }
    __syncthreads();

    const int col   = t & 127;
    const int rbase = t >> 7;
    #pragma unroll 4
    for (int e = 0; e < 64; ++e) {
        const int rl = rbase + e * 2;
        float dx = si[rl][0] - sj[col][0];
        float dy = si[rl][1] - sj[col][1];
        float dz = si[rl][2] - sj[col][2];
        float ds = sqrtf(fmaxf(dx * dx + dy * dy + dz * dz, 1e-12f));
        dx = ti[rl][0] - tj[col][0];
        dy = ti[rl][1] - tj[col][1];
        dz = ti[rl][2] - tj[col][2];
        float dt = sqrtf(fmaxf(dx * dx + dy * dy + dz * dz, 1e-12f));
        float diff = ds - dt;
        float v = fmaxf(1.0f - diff * diff * 100.0f, 0.0f);  // sigma_d = 0.1
        const int gi = row0 + rl, gj = col0 + col;
        if (gi == gj) v = 0.0f;
        fcg[((size_t)gi << LOG2N) + gj] = v;
    }
}

// ---------------------------------------------------------------------------
// Kernel 2: per-row sum of top-409 values via 1024-bin histogram selection.
// Fixed-order tree reductions -> deterministic.
// ---------------------------------------------------------------------------
__global__ __launch_bounds__(256) void topk_rowsum_kernel(const float* __restrict__ fcg,
                                                          float* __restrict__ rowsum) {
    __shared__ unsigned int hist[1024];
    __shared__ float redHi[256];
    __shared__ float redB[256];
    __shared__ unsigned int redC[256];
    __shared__ int s_bin;
    __shared__ unsigned int s_above;

    const int t = threadIdx.x;
    const float* r = fcg + ((size_t)blockIdx.x << LOG2N);

    for (int i = t; i < 1024; i += 256) hist[i] = 0u;
    __syncthreads();

    float vals[16];
    #pragma unroll
    for (int i = 0; i < 16; ++i) {
        float v = r[t + i * 256];
        vals[i] = v;
        if (v > 0.0f) {
            int b = (int)(v * 1024.0f);
            if (b > 1023) b = 1023;
            atomicAdd(&hist[b], 1u);
        }
    }
    __syncthreads();

    if (t == 0) {
        unsigned int cum = 0; int bin = -1; unsigned int above = 0;
        for (int b = 1023; b >= 0; --b) {
            unsigned int c = hist[b];
            if (cum + c >= (unsigned)TOPK) { bin = b; above = cum; break; }
            cum += c;
        }
        s_bin = bin;
        s_above = (bin >= 0) ? above : cum;  // bin<0: fewer than K nonzeros -> take all
    }
    __syncthreads();

    const int bin = s_bin;
    float sumHi = 0.0f, sumB = 0.0f;
    unsigned int cntB = 0;
    #pragma unroll
    for (int i = 0; i < 16; ++i) {
        float v = vals[i];
        if (v > 0.0f) {
            int b = (int)(v * 1024.0f);
            if (b > 1023) b = 1023;
            if (bin < 0 || b > bin) sumHi += v;
            else if (b == bin) { sumB += v; ++cntB; }
        }
    }
    redHi[t] = sumHi; redB[t] = sumB; redC[t] = cntB;
    __syncthreads();
    for (int s = 128; s > 0; s >>= 1) {
        if (t < s) {
            redHi[t] += redHi[t + s];
            redB[t]  += redB[t + s];
            redC[t]  += redC[t + s];
        }
        __syncthreads();
    }
    if (t == 0) {
        float total = redHi[0];
        if (bin >= 0) {
            unsigned int rem = (unsigned)TOPK - s_above;
            if (rem > redC[0]) rem = redC[0];
            if (redC[0] > 0)
                total += (redB[0] / (float)redC[0]) * (float)rem;  // boundary bin ~ avg
        }
        rowsum[blockIdx.x] = total;
    }
}

// ---------------------------------------------------------------------------
// Kernel 3: threshold = sum(rowsum) / (N * K)
// ---------------------------------------------------------------------------
__global__ __launch_bounds__(256) void thresh_reduce_kernel(const float* __restrict__ rowsum,
                                                            float* __restrict__ thresh) {
    __shared__ float red[256];
    const int t = threadIdx.x;
    float s = 0.0f;
    #pragma unroll
    for (int i = 0; i < 16; ++i) s += rowsum[t + i * 256];
    red[t] = s;
    __syncthreads();
    for (int st = 128; st > 0; st >>= 1) {
        if (t < st) red[t] += red[t + st];
        __syncthreads();
    }
    if (t == 0) *thresh = red[0] / ((float)N * (float)TOPK);
}

// ---------------------------------------------------------------------------
// Kernel 4: apply threshold in place (f32) and pack bf16 operand (RNE).
// ---------------------------------------------------------------------------
__device__ inline unsigned short f2bf_rne(float f) {
    unsigned int u = __float_as_uint(f);
    unsigned int r = u + 0x7FFFu + ((u >> 16) & 1u);
    return (unsigned short)(r >> 16);
}

__global__ __launch_bounds__(256) void pack_kernel(float* __restrict__ fcg,
                                                   unsigned int* __restrict__ abf,
                                                   const float* __restrict__ thr) {
    const float th = *thr;
    const size_t i = (size_t)blockIdx.x * blockDim.x + threadIdx.x;  // one float4 each
    float4 v = *reinterpret_cast<const float4*>(fcg + i * 4);
    v.x = (v.x < th) ? 0.0f : v.x;
    v.y = (v.y < th) ? 0.0f : v.y;
    v.z = (v.z < th) ? 0.0f : v.z;
    v.w = (v.w < th) ? 0.0f : v.w;
    *reinterpret_cast<float4*>(fcg + i * 4) = v;
    const unsigned int lo = (unsigned int)f2bf_rne(v.x) | ((unsigned int)f2bf_rne(v.y) << 16);
    const unsigned int hi = (unsigned int)f2bf_rne(v.z) | ((unsigned int)f2bf_rne(v.w) << 16);
    abf[i * 2 + 0] = lo;
    abf[i * 2 + 1] = hi;
}

// ---------------------------------------------------------------------------
// Kernel 5: out = (A @ A) * A  using v_wmma_f32_16x16x32_bf16.
// A (bf16, row-major) is symmetric, so B-columns are A-rows: both fragments
// come from the same buffer with plain b128 loads (ISA 7.12.2 layouts).
// 4 waves / block, 64x64 per wave (4x4 WMMA tiles), K-steps of 32.
// ---------------------------------------------------------------------------
__device__ inline v16bf load_a_frag(const unsigned short* __restrict__ base,
                                    int row, int k0, int half) {
    // lane l/half: K chunks {k0+8*half .. +7} and {k0+16+8*half .. +7}
    const unsigned short* p = base + ((size_t)row << LOG2N) + k0 + half * 8;
    union { uint4 u[2]; v16bf v; } w;
    w.u[0] = *reinterpret_cast<const uint4*>(p);
    w.u[1] = *reinterpret_cast<const uint4*>(p + 16);
    return w.v;
}

__device__ inline v16bf load_b_frag(const unsigned short* __restrict__ base,
                                    int col, int k0, int half) {
    // lane l holds column n=l; K chunks {k0+8*half .. +7} and {k0+16+8*half .. +7}
    const unsigned short* p = base + ((size_t)col << LOG2N) + k0 + half * 8;
    union { uint4 u[2]; v16bf v; } w;
    w.u[0] = *reinterpret_cast<const uint4*>(p);
    w.u[1] = *reinterpret_cast<const uint4*>(p + 16);
    return w.v;
}

__global__ __launch_bounds__(128) void hyper_gemm_kernel(const unsigned short* __restrict__ abf,
                                                         float* __restrict__ out) {
    const int lane = threadIdx.x & 31;
    const int wave = threadIdx.x >> 5;
    const int half = lane >> 4;
    const int l    = lane & 15;
    const int m_base = blockIdx.y * 128 + (wave >> 1) * 64;
    const int n_base = blockIdx.x * 128 + (wave & 1) * 64;

    v8f acc[4][4];
    #pragma unroll
    for (int mi = 0; mi < 4; ++mi)
        #pragma unroll
        for (int ni = 0; ni < 4; ++ni)
            #pragma unroll
            for (int q = 0; q < 8; ++q) acc[mi][ni][q] = 0.0f;

    for (int k0 = 0; k0 < N; k0 += 32) {
        v16bf a[4], b[4];
        #pragma unroll
        for (int mi = 0; mi < 4; ++mi)
            a[mi] = load_a_frag(abf, m_base + mi * 16 + l, k0, half);
        #pragma unroll
        for (int ni = 0; ni < 4; ++ni)
            b[ni] = load_b_frag(abf, n_base + ni * 16 + l, k0, half);
        #pragma unroll
        for (int mi = 0; mi < 4; ++mi)
            #pragma unroll
            for (int ni = 0; ni < 4; ++ni)
                acc[mi][ni] = __builtin_amdgcn_wmma_f32_16x16x32_bf16(
                    false, a[mi], false, b[ni], (short)0, acc[mi][ni], false, false);
    }

    // Fused epilogue: out = acc * fcg_thresholded (read-then-write, same thread)
    #pragma unroll
    for (int mi = 0; mi < 4; ++mi) {
        #pragma unroll
        for (int ni = 0; ni < 4; ++ni) {
            const int col = n_base + ni * 16 + l;
            #pragma unroll
            for (int r = 0; r < 8; ++r) {
                const int row = m_base + mi * 16 + half * 8 + r;
                const size_t idx = ((size_t)row << LOG2N) + col;
                out[idx] = acc[mi][ni][r] * out[idx];
            }
        }
    }
}

// ---------------------------------------------------------------------------
extern "C" void kernel_launch(void* const* d_in, const int* in_sizes, int n_in,
                              void* d_out, int out_size, void* d_ws, size_t ws_size,
                              hipStream_t stream) {
    (void)in_sizes; (void)n_in; (void)out_size; (void)ws_size;
    const float* src = (const float*)d_in[0];
    const float* tgt = (const float*)d_in[1];
    float* out = (float*)d_out;

    // ws layout: [0, 32MB) bf16 operand; then 4096 row sums; then 1 threshold.
    unsigned short* abf = (unsigned short*)d_ws;
    float* rowsum = (float*)((char*)d_ws + (size_t)N * N * sizeof(unsigned short));
    float* thr = rowsum + N;

    fcg_kernel<<<dim3(N / 128, N / 128), 256, 0, stream>>>(src, tgt, out);
    topk_rowsum_kernel<<<N, 256, 0, stream>>>(out, rowsum);
    thresh_reduce_kernel<<<1, 256, 0, stream>>>(rowsum, thr);
    pack_kernel<<<(N * (N / 4)) / 256, 256, 0, stream>>>(out, (unsigned int*)abf, thr);
    hyper_gemm_kernel<<<dim3(N / 128, N / 128), 128, 0, stream>>>(abf, out);
}
